// GAT_31988916421098
// MI455X (gfx1250) — matrix operand order repaired and checked by reading, not compiled, not run on hardware
//
#include <hip/hip_runtime.h>
#include <cstddef>

#define DIM 128
#define HEADS 8
#define DH 16
#define FF 512
#define LDH 136   // padded LDS leading dim (halves): 272B rows -> 16B aligned, conflict-free

typedef __attribute__((ext_vector_type(16))) __bf16 v16bf;
typedef __attribute__((ext_vector_type(8)))  float  v8f;
typedef __attribute__((ext_vector_type(4)))  unsigned int v4u;

union FragU { v16bf bf; v4u u2[2]; unsigned int u[8]; };

// native conversions -> backend v_cvt instructions
__device__ __forceinline__ unsigned short f2bf(float f) {
    return __builtin_bit_cast(unsigned short, (__bf16)f);
}
__device__ __forceinline__ unsigned int bfpack2(float a, float b) {
    return (unsigned int)f2bf(a) | ((unsigned int)f2bf(b) << 16);
}

// ---- A fragment (16x32 bf16) from a bf16 row (global or LDS), contiguous k order (same as load_bfrag):
// lanes 0-15 (row M=l): K = kc+0..15
// lanes 16-31 (row M=l): K = kc+16..31
// rowbase must already include row*ld for this lane's row.
__device__ __forceinline__ v16bf load_afrag(const unsigned short* rowbase, int kc) {
    const int hi = (threadIdx.x >> 4) & 1;
    const unsigned short* p = rowbase + kc + hi * 16;
    FragU f;
    f.u2[0] = *(const v4u*)p;
    f.u2[1] = *(const v4u*)(p + 8);
    return f.bf;
}

// ---- B fragment (32x16 bf16) from transposed bf16 weights WT[out][K]:
// lanes 0-15 (col N=l): K = kc+0..15 ; lanes 16-31 (col N=l): K = kc+16..31
__device__ __forceinline__ v16bf load_bfrag(const unsigned short* wt, int ldk, int n0, int kc) {
    const int lane = threadIdx.x & 31;
    const int l = lane & 15, hi = lane >> 4;
    const unsigned short* p = wt + (size_t)(n0 + l) * ldk + kc + hi * 16;
    FragU f;
    f.u2[0] = *(const v4u*)p;
    f.u2[1] = *(const v4u*)(p + 8);
    return f.bf;
}

__device__ __forceinline__ v8f wmma_bf16(v16bf a, v16bf b, v8f c) {
    return __builtin_amdgcn_wmma_f32_16x16x32_bf16(false, a, false, b, (short)0, c, false, false);
}

// ordered-uint encoding for float atomic max (memset 0 == "empty")
__device__ __forceinline__ unsigned int f2ord(float f) {
    unsigned int u = __float_as_uint(f);
    return (u & 0x80000000u) ? ~u : (u | 0x80000000u);
}
__device__ __forceinline__ float ord2f(unsigned int o) {
    unsigned int u = (o & 0x80000000u) ? (o & 0x7FFFFFFFu) : ~o;
    return __uint_as_float(u);
}

// ---------------- elementwise fp32 -> bf16 (8 elems/thread, vectorized) ----------------
__global__ void cvt_bf16_kernel(const float* __restrict__ x, unsigned short* __restrict__ y,
                                long long n8) {
    long long i = (long long)blockIdx.x * blockDim.x + threadIdx.x;
    if (i >= n8) return;
    const float4* xp = (const float4*)x + i * 2;
    float4 a = xp[0], b = xp[1];
    uint4 o;
    o.x = bfpack2(a.x, a.y); o.y = bfpack2(a.z, a.w);
    o.z = bfpack2(b.x, b.y); o.w = bfpack2(b.z, b.w);
    ((uint4*)y)[i] = o;
}

// ---------------- weight transpose + bf16 convert: dst[n*K+k] = src[k*Nc+n] ----------------
__global__ void prep_wt_kernel(const float* __restrict__ w, unsigned short* __restrict__ wt,
                               int K, int Nc) {
    int idx = blockIdx.x * blockDim.x + threadIdx.x;
    if (idx >= K * Nc) return;
    int nn = idx / K, kk = idx - nn * K;
    wt[idx] = f2bf(w[(size_t)kk * Nc + nn]);
}

// ---------------- fused Q/K/V projection (WMMA bf16) ----------------
__global__ __launch_bounds__(256) void qkv_kernel(
    const unsigned short* __restrict__ featb,
    const unsigned short* __restrict__ WqT, const unsigned short* __restrict__ WkT,
    const unsigned short* __restrict__ WvT,
    const float* __restrict__ bq, const float* __restrict__ bk, const float* __restrict__ bv,
    float* __restrict__ q, float* __restrict__ k, float* __restrict__ v, int n)
{
    const int tid = threadIdx.x, wave = tid >> 5, lane = tid & 31;
    const int l = lane & 15, hi = lane >> 4;
    const int r0 = blockIdx.x * 128;
    const int m0 = wave * 16;
    const bool full = (r0 + 128 <= n);          // uniform: fast-path stores

    __builtin_prefetch(WqT + tid * 64, 0, 3);   // weight tables: 96KB total, keep hot

    int arow = r0 + m0 + l; if (arow >= n) arow = n - 1;   // clamp; stores are guarded
    const unsigned short* rowbase = featb + (size_t)arow * DIM;
    v16bf a[4];
#pragma unroll
    for (int c = 0; c < 4; ++c) a[c] = load_afrag(rowbase, c * 32);

    const unsigned short* Ws[3] = {WqT, WkT, WvT};
    const float* Bs[3] = {bq, bk, bv};
    float* Os[3] = {q, k, v};
#pragma unroll
    for (int o = 0; o < 3; ++o) {
        const unsigned short* wt = Ws[o];
        const float* bias = Bs[o];
        float* out = Os[o];
#pragma unroll
        for (int nt = 0; nt < 8; ++nt) {
            const int n0 = nt * 16;
            float bc = bias[n0 + l];
            v8f acc = {bc, bc, bc, bc, bc, bc, bc, bc};
#pragma unroll
            for (int c = 0; c < 4; ++c)
                acc = wmma_bf16(a[c], load_bfrag(wt, DIM, n0, c * 32), acc);
            if (full) {
                float* p = out + (size_t)(r0 + m0 + hi * 8) * DIM + n0 + l;
#pragma unroll
                for (int r = 0; r < 8; ++r) p[(size_t)r * DIM] = acc[r];
            } else {
#pragma unroll
                for (int r = 0; r < 8; ++r) {
                    int row = r0 + m0 + r + hi * 8;
                    if (row < n) out[(size_t)row * DIM + n0 + l] = acc[r];
                }
            }
        }
    }
}

// ---------------- edge kernels (one thread per edge,head) ----------------
__global__ void edge_score_kernel(const float* __restrict__ q, const float* __restrict__ k,
                                  const int* __restrict__ src, const int* __restrict__ dst,
                                  float* __restrict__ e, unsigned int* __restrict__ emaxk, int E)
{
    long long t = (long long)blockIdx.x * blockDim.x + threadIdx.x;
    if (t >= (long long)E * HEADS) return;
    int h = (int)(t & (HEADS - 1));
    int eid = (int)(t >> 3);
    int s = src[eid], d = dst[eid];
    const float* kp = k + (size_t)s * DIM + h * DH;
    const float* qp = q + (size_t)d * DIM + h * DH;
    float acc = 0.f;
#pragma unroll
    for (int j = 0; j < 4; ++j) {
        float4 kv = ((const float4*)kp)[j];
        float4 qv = ((const float4*)qp)[j];
        acc += kv.x * qv.x + kv.y * qv.y + kv.z * qv.z + kv.w * qv.w;
    }
    acc *= 0.08838834764831845f;   // 1/sqrt(128)
    e[t] = acc;
    atomicMax(&emaxk[(size_t)d * HEADS + h], f2ord(acc));
}

__global__ void edge_exp_kernel(const float* __restrict__ ein, const int* __restrict__ dst,
                                const unsigned int* __restrict__ emaxk,
                                float* __restrict__ ex, float* __restrict__ denom, int E)
{
    long long t = (long long)blockIdx.x * blockDim.x + threadIdx.x;
    if (t >= (long long)E * HEADS) return;
    int h = (int)(t & (HEADS - 1));
    int eid = (int)(t >> 3);
    int d = dst[eid];
    float m = ord2f(emaxk[(size_t)d * HEADS + h]);
    float x = __expf(ein[t] - m);
    ex[t] = x;
    atomicAdd(&denom[(size_t)d * HEADS + h], x);
}

__global__ void edge_agg_kernel(const float* __restrict__ ex, const float* __restrict__ v,
                                const int* __restrict__ src, const int* __restrict__ dst,
                                const float* __restrict__ denom, float* __restrict__ agg, int E)
{
    long long t = (long long)blockIdx.x * blockDim.x + threadIdx.x;
    if (t >= (long long)E * HEADS) return;
    int h = (int)(t & (HEADS - 1));
    int eid = (int)(t >> 3);
    int s = src[eid], d = dst[eid];
    float a = ex[t] / denom[(size_t)d * HEADS + h];
    const float* vp = v + (size_t)s * DIM + h * DH;
    float* op = agg + (size_t)d * DIM + h * DH;
#pragma unroll
    for (int j = 0; j < DH; ++j) atomicAdd(&op[j], vp[j] * a);
}

// ---------------- layernorm: out = LN(x (+ y)); optional bf16 copy; one wave32 per row ----
__global__ __launch_bounds__(256) void ln_kernel(const float* __restrict__ x,
                                                 const float* __restrict__ y,
                                                 const float* __restrict__ g,
                                                 const float* __restrict__ b,
                                                 float* __restrict__ out,
                                                 unsigned short* __restrict__ outb, int n)
{
    const int wave = threadIdx.x >> 5, lane = threadIdx.x & 31;
    long long row = (long long)blockIdx.x * 8 + wave;
    if (row >= n) return;
    float4 s = *(const float4*)(x + row * DIM + lane * 4);
    if (y) {
        float4 yv = *(const float4*)(y + row * DIM + lane * 4);
        s.x += yv.x; s.y += yv.y; s.z += yv.z; s.w += yv.w;
    }
    float sum = s.x + s.y + s.z + s.w;
    float sq  = s.x * s.x + s.y * s.y + s.z * s.z + s.w * s.w;
#pragma unroll
    for (int off = 16; off > 0; off >>= 1) {
        sum += __shfl_xor(sum, off, 32);
        sq  += __shfl_xor(sq,  off, 32);
    }
    float mu  = sum * (1.0f / DIM);
    float var = sq * (1.0f / DIM) - mu * mu;
    float inv = rsqrtf(var + 1e-5f);
    float4 gv = *(const float4*)(g + lane * 4);
    float4 bv = *(const float4*)(b + lane * 4);
    float4 o;
    o.x = (s.x - mu) * inv * gv.x + bv.x;
    o.y = (s.y - mu) * inv * gv.y + bv.y;
    o.z = (s.z - mu) * inv * gv.z + bv.z;
    o.w = (s.w - mu) * inv * gv.w + bv.w;
    *(float4*)(out + row * DIM + lane * 4) = o;
    if (outb) {
        uint2 ob;
        ob.x = bfpack2(o.x, o.y);
        ob.y = bfpack2(o.z, o.w);
        *(uint2*)(outb + row * DIM + lane * 4) = ob;
    }
}

// ---------------- fused FFN: t = rst + (PReLU(rst@W1+b1))@W2 + b2 ----------------
__global__ __launch_bounds__(256) void ffn_kernel(
    const float* __restrict__ rst, const unsigned short* __restrict__ rstb,
    const unsigned short* __restrict__ W1T, const float* __restrict__ b1,
    const float* __restrict__ alpha,
    const unsigned short* __restrict__ W2T, const float* __restrict__ b2,
    float* __restrict__ t, int n)
{
    __shared__ unsigned short Hs[128 * LDH];   // bf16 hidden chunk, wave-private rows

    const int tid = threadIdx.x, wave = tid >> 5, lane = tid & 31;
    const int l = lane & 15, hi = lane >> 4;
    const int r0 = blockIdx.x * 128;
    const int m0 = wave * 16;
    const bool full = (r0 + 128 <= n);          // uniform: fast-path stores

    __builtin_prefetch(W1T + tid * 256, 0, 3);   // 128KB W1T
    __builtin_prefetch(W2T + tid * 256, 0, 3);   // 128KB W2T

    int arow = r0 + m0 + l; if (arow >= n) arow = n - 1;
    const unsigned short* rowbase = rstb + (size_t)arow * DIM;
    v16bf a1[4];
#pragma unroll
    for (int c = 0; c < 4; ++c) a1[c] = load_afrag(rowbase, c * 32);

    v8f acc2[8];
#pragma unroll
    for (int nt = 0; nt < 8; ++nt) {
        float bc = b2[nt * 16 + l];
        v8f z = {bc, bc, bc, bc, bc, bc, bc, bc};
        acc2[nt] = z;
    }

#pragma unroll
    for (int ch = 0; ch < 4; ++ch) {           // hidden columns ch*128 .. ch*128+127
        // GEMM1: this wave's 16-row strip of h chunk, + bias + PReLU -> bf16 LDS
#pragma unroll
        for (int nt = 0; nt < 8; ++nt) {
            const int ncol = ch * 128 + nt * 16 + l;
            float bc = b1[ncol];
            v8f acc = {bc, bc, bc, bc, bc, bc, bc, bc};
#pragma unroll
            for (int c = 0; c < 4; ++c)
                acc = wmma_bf16(a1[c], load_bfrag(W1T, DIM, ch * 128 + nt * 16, c * 32), acc);
            float al = alpha[ncol];
            unsigned short* hp = Hs + (m0 + hi * 8) * LDH + nt * 16 + l;
#pragma unroll
            for (int r = 0; r < 8; ++r) {
                float hv = acc[r];
                hv = hv > 0.f ? hv : al * hv;
                hp[r * LDH] = f2bf(hv);
            }
        }
        // GEMM2 partial accumulate over this K chunk (rows wave-private: no barrier needed)
#pragma unroll
        for (int c = 0; c < 4; ++c) {
            v16bf a2 = load_afrag(Hs + (m0 + l) * LDH, c * 32);
#pragma unroll
            for (int nt = 0; nt < 8; ++nt)
                acc2[nt] = wmma_bf16(a2, load_bfrag(W2T, FF, nt * 16, ch * 128 + c * 32), acc2[nt]);
        }
    }

    // epilogue: residual add (acc2 already includes b2)
    if (full) {
#pragma unroll
        for (int nt = 0; nt < 8; ++nt) {
            const size_t base = (size_t)(r0 + m0 + hi * 8) * DIM + nt * 16 + l;
#pragma unroll
            for (int r = 0; r < 8; ++r)
                t[base + (size_t)r * DIM] = rst[base + (size_t)r * DIM] + acc2[nt][r];
        }
    } else {
#pragma unroll
        for (int nt = 0; nt < 8; ++nt) {
#pragma unroll
            for (int r = 0; r < 8; ++r) {
                int row = r0 + m0 + r + hi * 8;
                if (row < n) {
                    size_t idx = (size_t)row * DIM + nt * 16 + l;
                    t[idx] = rst[idx] + acc2[nt][r];
                }
            }
        }
    }
}

extern "C" void kernel_launch(void* const* d_in, const int* in_sizes, int n_in,
                              void* d_out, int out_size, void* d_ws, size_t ws_size,
                              hipStream_t stream)
{
    const float* feat = (const float*)d_in[0];
    const int*   src  = (const int*)d_in[1];
    const int*   dst  = (const int*)d_in[2];
    const float* Wq   = (const float*)d_in[3];
    const float* bq   = (const float*)d_in[4];
    const float* Wk   = (const float*)d_in[5];
    const float* bk   = (const float*)d_in[6];
    const float* Wv   = (const float*)d_in[7];
    const float* bv   = (const float*)d_in[8];
    const float* ln_g = (const float*)d_in[9];
    const float* ln_b = (const float*)d_in[10];
    const float* W1   = (const float*)d_in[11];
    const float* b1   = (const float*)d_in[12];
    const float* alph = (const float*)d_in[13];
    const float* W2   = (const float*)d_in[14];
    const float* b2   = (const float*)d_in[15];

    const int n = in_sizes[0] / DIM;
    const int E = in_sizes[1];

    char* ws = (char*)d_ws;
    size_t off = 0;
    auto alloc = [&](size_t bytes) -> void* {
        void* p = ws + off;
        off = (off + bytes + 255) & ~(size_t)255;
        return p;
    };
    float* q      = (float*)alloc((size_t)n * DIM * 4);
    float* k      = (float*)alloc((size_t)n * DIM * 4);
    float* v      = (float*)alloc((size_t)n * DIM * 4);
    float* e      = (float*)alloc((size_t)E * HEADS * 4);
    unsigned int* emaxk = (unsigned int*)alloc((size_t)n * HEADS * 4);
    float* denom  = (float*)alloc((size_t)n * HEADS * 4);
    float* agg    = (float*)alloc((size_t)n * DIM * 4);   // reused as FFN output t
    float* rstln  = (float*)alloc((size_t)n * DIM * 4);
    unsigned short* featb  = (unsigned short*)alloc((size_t)n * DIM * 2);
    unsigned short* rstlnb = (unsigned short*)alloc((size_t)n * DIM * 2);
    unsigned short* WqT = (unsigned short*)alloc((size_t)DIM * DIM * 2);
    unsigned short* WkT = (unsigned short*)alloc((size_t)DIM * DIM * 2);
    unsigned short* WvT = (unsigned short*)alloc((size_t)DIM * DIM * 2);
    unsigned short* W1T = (unsigned short*)alloc((size_t)DIM * FF * 2);
    unsigned short* W2T = (unsigned short*)alloc((size_t)DIM * FF * 2);

    // weight prep (bf16 transposed copies) + activation bf16 copy
    prep_wt_kernel<<<(DIM * DIM + 255) / 256, 256, 0, stream>>>(Wq, WqT, DIM, DIM);
    prep_wt_kernel<<<(DIM * DIM + 255) / 256, 256, 0, stream>>>(Wk, WkT, DIM, DIM);
    prep_wt_kernel<<<(DIM * DIM + 255) / 256, 256, 0, stream>>>(Wv, WvT, DIM, DIM);
    prep_wt_kernel<<<(DIM * FF + 255) / 256, 256, 0, stream>>>(W1, W1T, DIM, FF);  // [512][128]
    prep_wt_kernel<<<(DIM * FF + 255) / 256, 256, 0, stream>>>(W2, W2T, FF, DIM);  // [128][512]

    const long long n8 = (long long)n * DIM / 8;
    cvt_bf16_kernel<<<(int)((n8 + 255) / 256), 256, 0, stream>>>(feat, featb, n8);

    hipMemsetAsync(emaxk, 0, (size_t)n * HEADS * 4, stream);   // ord-encoded "-inf"
    hipMemsetAsync(denom, 0, (size_t)n * HEADS * 4, stream);
    hipMemsetAsync(agg,   0, (size_t)n * DIM * 4, stream);

    const int nb = (n + 127) / 128;
    qkv_kernel<<<nb, 256, 0, stream>>>(featb, WqT, WkT, WvT, bq, bk, bv, q, k, v, n);

    const long long eh = (long long)E * HEADS;
    const int ebn = (int)((eh + 255) / 256);
    edge_score_kernel<<<ebn, 256, 0, stream>>>(q, k, src, dst, e, emaxk, E);
    edge_exp_kernel<<<ebn, 256, 0, stream>>>(e, dst, emaxk, e, denom, E);
    edge_agg_kernel<<<ebn, 256, 0, stream>>>(e, v, src, dst, denom, agg, E);

    ln_kernel<<<(n + 7) / 8, 256, 0, stream>>>(agg, feat, ln_g, ln_b, rstln, rstlnb, n);
    ffn_kernel<<<nb, 256, 0, stream>>>(rstln, rstlnb, W1T, b1, alph, W2T, b2, agg, n);
    ln_kernel<<<(n + 7) / 8, 256, 0, stream>>>(agg, nullptr, ln_g, ln_b, (float*)d_out, nullptr, n);
}
